// GIN_11390253269766
// MI455X (gfx1250) — matrix operand order repaired and checked by reading, not compiled, not run on hardware
//
#include <hip/hip_runtime.h>
#include <hip/hip_bf16.h>
#include <stdint.h>
#include <stddef.h>

// ---------------------------------------------------------------------------
// CDNA5 (gfx1250) GIN forward. bf16 WMMA (v_wmma_f32_16x16x32_bf16) GEMMs with
// fp32 accumulation, async global->LDS double-buffered B-tile staging
// (ASYNCcnt / s_wait_asynccnt), register-double-buffered B tiles, native
// f32->bf16 cvt for A, fused BN/ReLU in the A-load, fp32 atomic scatter-add
// aggregation, wave32 log-softmax.
// ---------------------------------------------------------------------------

typedef __attribute__((ext_vector_type(16))) __bf16 v16bf;
typedef __attribute__((ext_vector_type(8)))  float  v8f;

#define GEMM_BLOCK 128   // 4 waves per block

#if defined(__HIP_DEVICE_COMPILE__) && \
    __has_builtin(__builtin_amdgcn_global_load_async_to_lds_b128)
#define USE_ASYNC_LDS 1
#else
#define USE_ASYNC_LDS 0
#endif

#if USE_ASYNC_LDS
// The builtin's params are pointers to GCC-style v4i in AS1 (global) / AS3 (LDS).
typedef int g_v4i __attribute__((vector_size(16)));
__device__ __forceinline__ void async_copy_b128(const uint4* gsrc, uint4* ldst) {
  __builtin_amdgcn_global_load_async_to_lds_b128(
      (__attribute__((address_space(1))) g_v4i*)gsrc,
      (__attribute__((address_space(3))) g_v4i*)ldst, 0, 0);
}
#endif

union Pack16 {
  uint32_t d[8];
  uint4    q[2];
  v16bf    v;
};

// Integer RNE f32->bf16 pair pack (used only in the tiny weight-pack kernel).
__device__ __forceinline__ uint32_t f2bf2_int(float lo, float hi) {
  union { float f; uint32_t u; } a, b;
  a.f = lo; b.f = hi;
  uint32_t ua = (a.u + 0x7FFFu + ((a.u >> 16) & 1u)) >> 16;   // RNE
  uint32_t ub = (b.u + 0x7FFFu + ((b.u >> 16) & 1u)) >> 16;
  return (ua & 0xFFFFu) | (ub << 16);
}

__device__ __forceinline__ void wait_asynccnt0() {
#if USE_ASYNC_LDS
#if __has_builtin(__builtin_amdgcn_s_wait_asynccnt)
  __builtin_amdgcn_s_wait_asynccnt(0);
#else
  asm volatile("s_wait_asynccnt 0" ::: "memory");
#endif
#endif
}

// ---------------------------------------------------------------------------
// Weight packer: fp32 W (K x NC, row-major) -> bf16 tiles in exact WMMA-B
// register layout. Tile (kt,nt) covers K=[32kt,32kt+32), N=[16nt,16nt+16).
// Lane l of a tile holds column n = 16nt + (l&15), K = 32kt+8*(l>>4)+{0..7, 16..23},
// 16 bf16 values packed pairwise into 8 dwords stored contiguously (2x b128).
// ---------------------------------------------------------------------------
__global__ void pack_w_kernel(const float* __restrict__ W, int K, int NC, int NCT,
                              uint32_t* __restrict__ out) {
  int idx = blockIdx.x * blockDim.x + threadIdx.x;         // one thread per (tile,lane)
  int total = (K >> 5) * NCT * 32;
  if (idx >= total) return;
  int lane = idx & 31;
  int tile = idx >> 5;
  int kt = tile / NCT, nt = tile % NCT;
  int n    = nt * 16 + (lane & 15);
  int koff = kt * 32 + ((lane >> 4) << 3);
  uint32_t d[8];
#pragma unroll
  for (int j = 0; j < 8; ++j) {
    float v0 = (n < NC) ? W[(size_t)(koff + 2 * j + ((j >> 2) << 3))     * NC + n] : 0.f;
    float v1 = (n < NC) ? W[(size_t)(koff + 2 * j + ((j >> 2) << 3) + 1) * NC + n] : 0.f;
    d[j] = f2bf2_int(v0, v1);
  }
  uint4* o = (uint4*)(out + (size_t)idx * 8);
  o[0] = make_uint4(d[0], d[1], d[2], d[3]);
  o[1] = make_uint4(d[4], d[5], d[6], d[7]);
}

// ---------------------------------------------------------------------------
// Wave-tiled WMMA GEMM:  Y(N x NC) = op(A)(N x K) @ Wpack + bias
// One wave computes a 16-row strip across NCT 16-col tiles (A read once).
// B tiles for each K-step are staged into LDS with async global->LDS loads,
// double-buffered so the next stage overlaps the current WMMA burst; within
// the burst, B register tiles are double-buffered to hide ds_load latency.
// AFFINE_A: a = relu(a*scaleA[k] + shiftA[k])  (fused BatchNorm+ReLU)
// RELU_OUT: y = relu(y)
// ---------------------------------------------------------------------------
template <int NCT, bool AFFINE_A, bool RELU_OUT>
__global__ __launch_bounds__(GEMM_BLOCK)
void gemm_wmma_kernel(const float* __restrict__ A, const uint32_t* __restrict__ Bp,
                      const float* __restrict__ bias,
                      const float* __restrict__ scaleA, const float* __restrict__ shiftA,
                      float* __restrict__ Y, int N, int K, int ldY, int NC) {
  extern __shared__ uint32_t smemB[];                // 2 * NCT * 256 dwords
  const int tid  = threadIdx.x;
  const int lane = tid & 31;
  const int wave = blockIdx.x * (GEMM_BLOCK >> 5) + (tid >> 5);

  int  row0   = wave << 4;
  bool active = row0 < N;
  if (row0 > N - 16) row0 = N - 16;                  // keep loads in-bounds; no early return
                                                     // (all threads must reach barriers)
  const int m  = lane & 15;
  const int hk = (lane >> 4) << 3;                   // 0 or 8: per-lane K sub-group (ISA 7.12.2)

  v8f acc[NCT] = {};
  const float* Arow = A + (size_t)(row0 + m) * (size_t)K;
  const int KT = K >> 5;

#if USE_ASYNC_LDS
  auto stage = [&](int kt, int buf) {
    const uint4* src = (const uint4*)Bp + (size_t)kt * NCT * 64;
    uint4*       dst = (uint4*)(smemB + (size_t)buf * NCT * 256);
    for (int i = tid; i < NCT * 64; i += GEMM_BLOCK) {
      async_copy_b128(src + i, dst + i);
    }
  };
  stage(0, 0);
#endif

  for (int kt = 0; kt < KT; ++kt) {
#if USE_ASYNC_LDS
    wait_asynccnt0();                                // buffer (kt&1) resident
    __syncthreads();
    if (kt + 1 < KT) stage(kt + 1, (kt + 1) & 1);    // prefetch next while computing
#endif
    const int kb = kt << 5;
    // A: 16 fp32 -> bf16 (native v_cvt); K set = {kb+hk..+7} U {kb+16+hk..+7}
    const float4* p0 = (const float4*)(Arow + kb + hk);
    const float4* p1 = (const float4*)(Arow + kb + 16 + hk);
    float4 x0 = p0[0], x1 = p0[1], x2 = p1[0], x3 = p1[1];
    float af[16] = { x0.x, x0.y, x0.z, x0.w, x1.x, x1.y, x1.z, x1.w,
                     x2.x, x2.y, x2.z, x2.w, x3.x, x3.y, x3.z, x3.w };
    if constexpr (AFFINE_A) {
#pragma unroll
      for (int j = 0; j < 8; ++j) {
        int k0 = kb + hk + j;
        int k1 = kb + 16 + hk + j;
        af[j]     = fmaxf(fmaf(af[j],     scaleA[k0], shiftA[k0]), 0.f);
        af[8 + j] = fmaxf(fmaf(af[8 + j], scaleA[k1], shiftA[k1]), 0.f);
      }
    }
    v16bf av;
#pragma unroll
    for (int j = 0; j < 16; ++j) av[j] = (__bf16)af[j];

#if USE_ASYNC_LDS
    const uint4* bkt = (const uint4*)(smemB + (size_t)(kt & 1) * NCT * 256) + lane * 2;
#else
    const uint4* bkt = (const uint4*)Bp + ((size_t)kt * NCT * 32 + lane) * 2;
#endif
    // Register-double-buffered B tiles: fetch nt+1 while WMMA consumes nt.
    Pack16 bp[2];
    bp[0].q[0] = bkt[0];
    bp[0].q[1] = bkt[1];
#pragma unroll
    for (int nt = 0; nt < NCT; ++nt) {
      const int cur = nt & 1;
      if (nt + 1 < NCT) {
        bp[cur ^ 1].q[0] = bkt[(nt + 1) * 64];
        bp[cur ^ 1].q[1] = bkt[(nt + 1) * 64 + 1];
      }
      acc[nt] = __builtin_amdgcn_wmma_f32_16x16x32_bf16(
          false, av, false, bp[cur].v, (short)0, acc[nt], false, false);
    }
  }

  // Epilogue: C/D layout -> row = row0 + j + 8*(lane>>4), col = 16nt + (lane&15)
  const int rup = (lane >> 4) << 3;
#pragma unroll
  for (int nt = 0; nt < NCT; ++nt) {
    int col = nt * 16 + (lane & 15);
    bool cok = (col < NC) && active;
    float bv = (col < NC) ? bias[col] : 0.f;
#pragma unroll
    for (int j = 0; j < 8; ++j) {
      float v = acc[nt][j] + bv;
      if (RELU_OUT) v = fmaxf(v, 0.f);
      if (cok) Y[(size_t)(row0 + rup + j) * ldY + col] = v;
    }
  }
}

// ---------------------------------------------------------------------------
// Graph aggregation: agg[:NT] = x[:NT]; then agg[tgt[e]] += x[src[e]]
// ---------------------------------------------------------------------------
__global__ void copy_kernel(float* __restrict__ dst, const float* __restrict__ src, int n) {
  int i = blockIdx.x * blockDim.x + threadIdx.x;
  if (i < n) dst[i] = src[i];
}

__global__ void zero_kernel(float* __restrict__ p, int n) {
  int i = blockIdx.x * blockDim.x + threadIdx.x;
  if (i < n) p[i] = 0.f;
}

__global__ void scatter_add_kernel(const float* __restrict__ x, const int* __restrict__ src,
                                   const int* __restrict__ tgt, int E, int K,
                                   float* __restrict__ agg) {
  int e = blockIdx.x;
  int c = threadIdx.x;                 // blockDim.x == K (128 or 256)
  if (e >= E) return;
  int s = src[e], t = tgt[e];
  atomicAdd(&agg[(size_t)t * K + c], x[(size_t)s * K + c]);
}

// ---------------------------------------------------------------------------
// BatchNorm: column sums/sumsq reduction, then scale/shift
// ---------------------------------------------------------------------------
__global__ void bn_stats_kernel(const float* __restrict__ h, int N, int C,
                                float* __restrict__ sums) {
  int c = threadIdx.x;                 // blockDim.x == C (256)
  float s = 0.f, s2 = 0.f;
  for (int r = blockIdx.x; r < N; r += gridDim.x) {
    float v = h[(size_t)r * C + c];
    s += v; s2 += v * v;
  }
  atomicAdd(&sums[c], s);
  atomicAdd(&sums[C + c], s2);
}

__global__ void bn_finalize_kernel(const float* __restrict__ sums,
                                   const float* __restrict__ g, const float* __restrict__ be,
                                   int C, float invN,
                                   float* __restrict__ scale, float* __restrict__ shift) {
  int c = blockIdx.x * blockDim.x + threadIdx.x;
  if (c >= C) return;
  float mu  = sums[c] * invN;
  float var = sums[C + c] * invN - mu * mu;
  float sc  = g[c] * rsqrtf(var + 1e-5f);
  scale[c] = sc;
  shift[c] = be[c] - mu * sc;
}

// ---------------------------------------------------------------------------
// Row-wise log-softmax, one wave32 per row
// ---------------------------------------------------------------------------
__global__ void log_softmax_kernel(const float* __restrict__ logits, int ld,
                                   int rows, int C, float* __restrict__ out) {
  int row  = blockIdx.x * (blockDim.x >> 5) + (threadIdx.x >> 5);
  int lane = threadIdx.x & 31;
  if (row >= rows) return;
  const float* p = logits + (size_t)row * ld;
  float m = -3.4e38f;
  for (int c = lane; c < C; c += 32) m = fmaxf(m, p[c]);
#pragma unroll
  for (int off = 16; off; off >>= 1) m = fmaxf(m, __shfl_xor(m, off, 32));
  float s = 0.f;
  for (int c = lane; c < C; c += 32) s += __expf(p[c] - m);
#pragma unroll
  for (int off = 16; off; off >>= 1) s += __shfl_xor(s, off, 32);
  float lse = m + __logf(s);
  for (int c = lane; c < C; c += 32) out[(size_t)row * C + c] = p[c] - lse;
}

// ---------------------------------------------------------------------------
// Host orchestration
// ---------------------------------------------------------------------------
static inline int cdiv(long long a, long long b) { return (int)((a + b - 1) / b); }

extern "C" void kernel_launch(void* const* d_in, const int* in_sizes, int n_in,
                              void* d_out, int out_size, void* d_ws, size_t ws_size,
                              hipStream_t stream) {
  (void)n_in; (void)out_size; (void)ws_size;
  constexpr int NT0 = 123904, NT1 = 11264, NT2 = 1024;
  constexpr int IN_CH = 128, HID = 256, OUT_CH = 47;

  const float* x   = (const float*)d_in[0];
  const int*   ei0 = (const int*)d_in[1];
  const int*   ei1 = (const int*)d_in[2];
  const int*   ei2 = (const int*)d_in[3];
  const int E0 = in_sizes[1] / 2, E1 = in_sizes[2] / 2, E2 = in_sizes[3] / 2;

  const float* cw1[3] = { (const float*)d_in[4],  (const float*)d_in[10], (const float*)d_in[16] };
  const float* cb1[3] = { (const float*)d_in[5],  (const float*)d_in[11], (const float*)d_in[17] };
  const float* cg [3] = { (const float*)d_in[6],  (const float*)d_in[12], (const float*)d_in[18] };
  const float* cbe[3] = { (const float*)d_in[7],  (const float*)d_in[13], (const float*)d_in[19] };
  const float* cw2[3] = { (const float*)d_in[8],  (const float*)d_in[14], (const float*)d_in[20] };
  const float* cb2[3] = { (const float*)d_in[9],  (const float*)d_in[15], (const float*)d_in[21] };
  const float* lin1_w = (const float*)d_in[22];
  const float* lin1_b = (const float*)d_in[23];
  const float* lin2_w = (const float*)d_in[24];
  const float* lin2_b = (const float*)d_in[25];

  // --- workspace carve-out ---
  char* wp = (char*)d_ws;
  auto alloc = [&](size_t bytes) -> void* {
    void* p = wp; wp += (bytes + 255) & ~(size_t)255; return p;
  };
  float* h_tmp = (float*)alloc((size_t)NT0 * HID * 4);    // GEMM1 output (per layer)
  float* hA    = (float*)alloc((size_t)NT0 * HID * 4);    // layer0 out / layer2 out
  float* hB    = (float*)alloc((size_t)NT1 * HID * 4);    // layer1 out / lin1 out
  float* agg   = (float*)alloc((size_t)NT0 * IN_CH * 4);  // aggregation buffer (max)
  float* bnsum = (float*)alloc(512 * 4);
  float* bnsc  = (float*)alloc(256 * 4);
  float* bnsh  = (float*)alloc(256 * 4);
  uint32_t* p_w1[3], *p_w2[3], *p_l1, *p_l2;
  for (int i = 0; i < 3; ++i) {
    int K = (i == 0) ? IN_CH : HID;
    p_w1[i] = (uint32_t*)alloc((size_t)K * 16 * 8 * 4);
    p_w2[i] = (uint32_t*)alloc((size_t)HID * 16 * 8 * 4);
  }
  p_l1 = (uint32_t*)alloc((size_t)HID * 16 * 8 * 4);
  p_l2 = (uint32_t*)alloc((size_t)HID * 3 * 8 * 4);
  float* logits = (float*)alloc((size_t)NT2 * 48 * 4);

  // --- pack all weights into WMMA-B bf16 layout ---
  auto pack = [&](const float* W, int K, int NC, int NCT, uint32_t* out) {
    int threads = (K >> 5) * NCT * 32;
    hipLaunchKernelGGL(pack_w_kernel, dim3(cdiv(threads, 256)), dim3(256), 0, stream,
                       W, K, NC, NCT, out);
  };
  for (int i = 0; i < 3; ++i) {
    pack(cw1[i], (i == 0) ? IN_CH : HID, HID, 16, p_w1[i]);
    pack(cw2[i], HID, HID, 16, p_w2[i]);
  }
  pack(lin1_w, HID, HID, 16, p_l1);
  pack(lin2_w, HID, OUT_CH, 3, p_l2);

  constexpr uint32_t SMEM16 = 2u * 16 * 256 * 4;   // double-buffered B, NCT=16: 32 KB
  constexpr uint32_t SMEM3  = 2u * 3  * 256 * 4;   // NCT=3: 6 KB

  auto gemm_plain = [&](const float* A, const uint32_t* Bp, const float* bias,
                        float* Y, int N, int K, bool relu) {
    int blocks = cdiv(N / 16, 4);
    if (relu)
      hipLaunchKernelGGL((gemm_wmma_kernel<16, false, true>), dim3(blocks), dim3(GEMM_BLOCK),
                         SMEM16, stream, A, Bp, bias, (const float*)nullptr,
                         (const float*)nullptr, Y, N, K, HID, HID);
    else
      hipLaunchKernelGGL((gemm_wmma_kernel<16, false, false>), dim3(blocks), dim3(GEMM_BLOCK),
                         SMEM16, stream, A, Bp, bias, (const float*)nullptr,
                         (const float*)nullptr, Y, N, K, HID, HID);
  };

  // --- one GIN layer ---
  auto gin_layer = [&](const float* xin, int Kin, const int* ei, int E, int NT, int li,
                       float* out /* NT x HID */) {
    const int* src = ei;
    const int* tgt = ei + E;
    // agg = xin[:NT] + segment_sum(xin[src], tgt)
    int ncopy = NT * Kin;
    hipLaunchKernelGGL(copy_kernel, dim3(cdiv(ncopy, 256)), dim3(256), 0, stream,
                       agg, xin, ncopy);
    hipLaunchKernelGGL(scatter_add_kernel, dim3(E), dim3(Kin), 0, stream,
                       xin, src, tgt, E, Kin, agg);
    // h = agg @ w1 + b1
    gemm_plain(agg, p_w1[li], cb1[li], h_tmp, NT, Kin, false);
    // batchnorm stats
    hipLaunchKernelGGL(zero_kernel, dim3(2), dim3(256), 0, stream, bnsum, 512);
    hipLaunchKernelGGL(bn_stats_kernel, dim3(256), dim3(HID), 0, stream,
                       h_tmp, NT, HID, bnsum);
    hipLaunchKernelGGL(bn_finalize_kernel, dim3(1), dim3(HID), 0, stream,
                       bnsum, cg[li], cbe[li], HID, 1.0f / (float)NT, bnsc, bnsh);
    // out = relu( relu(BN(h)) @ w2 + b2 )  — BN+ReLU fused into A-load
    int blocks = cdiv(NT / 16, 4);
    hipLaunchKernelGGL((gemm_wmma_kernel<16, true, true>), dim3(blocks), dim3(GEMM_BLOCK),
                       SMEM16, stream, h_tmp, p_w2[li], cb2[li], bnsc, bnsh,
                       out, NT, HID, HID, HID);
  };

  gin_layer(x,  IN_CH, ei0, E0, NT0, 0, hA);   // 1362944x128 -> 123904x256
  gin_layer(hA, HID,   ei1, E1, NT1, 1, hB);   // -> 11264x256
  gin_layer(hB, HID,   ei2, E2, NT2, 2, hA);   // -> 1024x256 (reuse hA)

  // lin1 + relu -> hB (1024x256)
  gemm_plain(hA, p_l1, lin1_b, hB, NT2, HID, true);

  // lin2 -> logits (1024 x 47, ld=48)
  hipLaunchKernelGGL((gemm_wmma_kernel<3, false, false>), dim3(cdiv(NT2 / 16, 4)),
                     dim3(GEMM_BLOCK), SMEM3, stream, hB, p_l2, lin2_b,
                     (const float*)nullptr, (const float*)nullptr,
                     logits, NT2, HID, 48, OUT_CH);

  // log_softmax -> d_out (1024 x 47)
  hipLaunchKernelGGL(log_softmax_kernel, dim3(cdiv(NT2, 8)), dim3(256), 0, stream,
                     logits, 48, NT2, OUT_CH, (float*)d_out);
}
